// GCNLayer_9010841387499
// MI455X (gfx1250) — matrix operand order repaired and checked by reading, not compiled, not run on hardware
//
#include <hip/hip_runtime.h>
#include <stdint.h>

#define N_NODES 50000
#define N_EDGES 400000
#define D       256
#define KEEP_P  0.9f

typedef __attribute__((ext_vector_type(16))) __bf16 v16bf;
typedef __attribute__((ext_vector_type(8)))  float  v8f;

union frag16 { v16bf v; uint4 q[2]; };   // 32-byte bf16 fragment loader

// ---------------------------------------------------------------------------
// Kernel 1: zero the degree arrays and the aggregation buffer (d_out).
// ---------------------------------------------------------------------------
__global__ void k_zero(float* __restrict__ degs /* 2*N floats */,
                       float* __restrict__ agg  /* N*D floats */) {
    const size_t total = (size_t)N_NODES * D;
    const size_t stride = (size_t)gridDim.x * blockDim.x;
    for (size_t i = (size_t)blockIdx.x * blockDim.x + threadIdx.x; i < total; i += stride) {
        agg[i] = 0.0f;
        if (i < 2u * N_NODES) degs[i] = 0.0f;
    }
}

// ---------------------------------------------------------------------------
// Kernel 2: degree counting from the edge list.
// ---------------------------------------------------------------------------
__global__ void k_degrees(const int* __restrict__ src, const int* __restrict__ dst,
                          float* __restrict__ deg_out, float* __restrict__ deg_in) {
    const int stride = gridDim.x * blockDim.x;
    for (int e = blockIdx.x * blockDim.x + threadIdx.x; e < N_EDGES; e += stride) {
        atomicAdd(&deg_out[src[e]], 1.0f);
        atomicAdd(&deg_in[dst[e]], 1.0f);
    }
}

// ---------------------------------------------------------------------------
// Kernel 3a: one-shot fp32 -> bf16 conversion of W (amortized over 3125 tiles).
// Row-major bf16 W is exactly the WMMA B-fragment layout.
// ---------------------------------------------------------------------------
__global__ void k_cvt_w(const float* __restrict__ W, __bf16* __restrict__ Wb) {
    const int i = (blockIdx.x * blockDim.x + threadIdx.x) * 4;
    if (i >= D * D) return;
    const float4 f = *(const float4*)(W + i);
    __bf16 o[4] = { (__bf16)f.x, (__bf16)f.y, (__bf16)f.z, (__bf16)f.w };
    *(uint2*)(Wb + i) = *(uint2*)o;
}

// ---------------------------------------------------------------------------
// Kernel 3b: fused row-normalize (rsqrt of out-degree) + fp32 -> bf16 of feat.
// ---------------------------------------------------------------------------
__global__ void k_cvt_feat(const float* __restrict__ feat,
                           const float* __restrict__ deg_out,
                           __bf16* __restrict__ fb) {
    const size_t i = ((size_t)blockIdx.x * blockDim.x + threadIdx.x) * 4;
    if (i >= (size_t)N_NODES * D) return;
    const int n = (int)(i >> 8);
    const float dg = deg_out[n];
    const float s  = rsqrtf(dg < 1.0f ? 1.0f : dg);
    const float4 f = *(const float4*)(feat + i);
    __bf16 o[4] = { (__bf16)(f.x * s), (__bf16)(f.y * s),
                    (__bf16)(f.z * s), (__bf16)(f.w * s) };
    *(uint2*)(fb + i) = *(uint2*)o;
}

// ---------------------------------------------------------------------------
// Kernel 3c: x = featbf @ Wbf via bf16 WMMA, f32 accum.
// One 16(M) x 64(N) strip per wave, 4 accumulators sharing one A fragment.
// Inner loop is pure loads + 4 WMMAs: no conversion VALU.
// ---------------------------------------------------------------------------
#define N_WAVETILES ((N_NODES / 16) * 4)   // 12500 waves total

__global__ __launch_bounds__(256) void k_gemm_wmma(
        const __bf16* __restrict__ fb, const __bf16* __restrict__ Wb,
        float* __restrict__ x) {
    const int lane  = threadIdx.x & 31;
    const int wave  = threadIdx.x >> 5;
    const int wid   = blockIdx.x * 8 + wave;      // 0 .. 12499 (+ tail waves)
    if (wid >= N_WAVETILES) return;               // whole-wave retire: EXEC stays all-1s
    const int mtile = wid >> 2;                   // 0 .. 3124
    const int nq    = wid & 3;                    // N-quad: 4 tiles of 16
    const int g     = lane >> 4;                  // lane group (0/1)
    const int lm    = lane & 15;

    const int m     = mtile * 16 + lm;            // A row owned by this lane
    const int nBase = nq * 64;

    const __bf16* __restrict__ arow = fb + (size_t)m * D;

    v8f c0 = {}, c1 = {}, c2 = {}, c3 = {};
#pragma unroll
    for (int kk = 0; kk < D; kk += 32) {
        // --- A fragment: 16x32 bf16. Lane group g holds K = g*8..g*8+7 in
        // VGPR0-3 and K = 16+g*8..16+g*8+7 in VGPR4-7 (ISA 16-bit A layout).
        // Each 8-element run is 16 bytes, 16B-aligned -> one b128 load.
        const int k0 = kk + g * 8;
        frag16 a;
        a.q[0] = *(const uint4*)(arow + k0);
        a.q[1] = *(const uint4*)(arow + k0 + 16);

        // --- B fragments: 4 adjacent 32x16 tiles. Lane l holds column N = lm of
        // each tile, K in the same order as A (g*8.., 16+g*8..).
        const __bf16* __restrict__ bcol =
            Wb + (size_t)(kk + g * 8) * D + nBase + lm;
        frag16 b0, b1, b2, b3;
#pragma unroll
        for (int s = 0; s < 16; ++s) {
            const size_t kr = (size_t)(s < 8 ? s : s + 8) * D;
            b0.v[s] = bcol[kr +  0]; b1.v[s] = bcol[kr + 16];
            b2.v[s] = bcol[kr + 32]; b3.v[s] = bcol[kr + 48];
        }

        c0 = __builtin_amdgcn_wmma_f32_16x16x32_bf16(false, a.v, false, b0.v, (short)0, c0, false, false);
        c1 = __builtin_amdgcn_wmma_f32_16x16x32_bf16(false, a.v, false, b1.v, (short)0, c1, false, false);
        c2 = __builtin_amdgcn_wmma_f32_16x16x32_bf16(false, a.v, false, b2.v, (short)0, c2, false, false);
        c3 = __builtin_amdgcn_wmma_f32_16x16x32_bf16(false, a.v, false, b3.v, (short)0, c3, false, false);
    }

    // --- C/D layout: col N = lane&15, row M = r + 8*(lane>>4) for VGPR r.
    float* __restrict__ orow = x + (size_t)(mtile * 16 + 8 * g) * D + nBase + lm;
#pragma unroll
    for (int r = 0; r < 8; ++r) {
        orow[(size_t)r * D +  0] = c0[r];
        orow[(size_t)r * D + 16] = c1[r];
        orow[(size_t)r * D + 32] = c2[r];
        orow[(size_t)r * D + 48] = c3[r];
    }
}

// ---------------------------------------------------------------------------
// Kernel 4: edge scatter-add.  One wave per edge; each lane moves 8 floats
// (two float4 loads from L2-resident x, 8 float atomics into agg[dst]).
// ---------------------------------------------------------------------------
__global__ __launch_bounds__(256) void k_scatter(
        const float* __restrict__ x, const int* __restrict__ src,
        const int* __restrict__ dst, float* __restrict__ agg) {
    const int gid  = blockIdx.x * blockDim.x + threadIdx.x;
    const int e    = gid >> 5;
    if (e >= N_EDGES) return;
    const int lane = gid & 31;

    const int sN = src[e];
    const int dN = dst[e];

    const float4* __restrict__ xs = (const float4*)(x + (size_t)sN * D + lane * 8);
    float* __restrict__ od = agg + (size_t)dN * D + lane * 8;

    const float4 v0 = xs[0];
    const float4 v1 = xs[1];
    atomicAdd(od + 0, v0.x); atomicAdd(od + 1, v0.y);
    atomicAdd(od + 2, v0.z); atomicAdd(od + 3, v0.w);
    atomicAdd(od + 4, v1.x); atomicAdd(od + 5, v1.y);
    atomicAdd(od + 6, v1.z); atomicAdd(od + 7, v1.w);
}

// ---------------------------------------------------------------------------
// Kernel 5: h = relu(agg * rsqrt(max(deg_in,1)) + b), then deterministic
// inverted dropout (p = 0.1) via an integer hash of the element index.
// ---------------------------------------------------------------------------
__device__ __forceinline__ uint32_t hash_u32(uint32_t v) {
    v ^= v >> 16; v *= 0x7feb352dU;
    v ^= v >> 15; v *= 0x846ca68bU;
    v ^= v >> 16;
    return v;
}

__global__ void k_finalize(float* __restrict__ out, const float* __restrict__ deg_in,
                           const float* __restrict__ bias) {
    const size_t total   = (size_t)N_NODES * D;
    const size_t stride  = (size_t)gridDim.x * blockDim.x;
    const uint32_t drop_thresh = 429496729u; // 0.1 * 2^32
    for (size_t i = (size_t)blockIdx.x * blockDim.x + threadIdx.x; i < total; i += stride) {
        const int n = (int)(i >> 8);
        const int c = (int)(i & 255);
        const float dg = deg_in[n];
        const float sc = rsqrtf(dg < 1.0f ? 1.0f : dg);
        float h = out[i] * sc + bias[c];
        h = h > 0.0f ? h : 0.0f;
        const uint32_t r = hash_u32((uint32_t)i ^ 0x9e3779b9u);
        out[i] = (r >= drop_thresh) ? h * (1.0f / KEEP_P) : 0.0f;
    }
}

// ---------------------------------------------------------------------------
// Launch. Inputs: feat[f32 N*D], W[f32 D*D], b[f32 D], src[i32 E], dst[i32 E].
// d_out: f32 N*D.
// Workspace layout (floats unless noted):
//   deg_out[N] | deg_in[N] | x[N*D] | Wbf[D*D bf16] | featbf[N*D bf16]
// total ~= 0.4 MB + 51.2 MB + 0.128 MB + 25.6 MB ~= 77.3 MB.
// ---------------------------------------------------------------------------
extern "C" void kernel_launch(void* const* d_in, const int* in_sizes, int n_in,
                              void* d_out, int out_size, void* d_ws, size_t ws_size,
                              hipStream_t stream) {
    const float* feat = (const float*)d_in[0];
    const float* W    = (const float*)d_in[1];
    const float* bias = (const float*)d_in[2];
    const int*   src  = (const int*)d_in[3];
    const int*   dst  = (const int*)d_in[4];
    float*       out  = (float*)d_out;

    float*  deg_out = (float*)d_ws;
    float*  deg_in  = deg_out + N_NODES;
    float*  x       = deg_in + N_NODES;                 // N*D f32
    __bf16* Wb      = (__bf16*)(x + (size_t)N_NODES * D); // D*D bf16
    __bf16* fb      = Wb + (size_t)D * D;               // N*D bf16

    // 1. zero degrees + aggregation buffer (d_out)
    k_zero<<<4096, 256, 0, stream>>>(deg_out, out);

    // 2. degree counts
    k_degrees<<<(N_EDGES + 255) / 256, 256, 0, stream>>>(src, dst, deg_out, deg_in);

    // 3a. W -> bf16 (once; reused by every M-tile)
    k_cvt_w<<<(D * D / 4 + 255) / 256, 256, 0, stream>>>(W, Wb);

    // 3b. feat * rsqrt(deg_out) -> bf16
    k_cvt_feat<<<((N_NODES * D / 4) + 255) / 256, 256, 0, stream>>>(feat, deg_out, fb);

    // 3c. projection via WMMA: 12500 waves, 8 waves/block
    k_gemm_wmma<<<(N_WAVETILES + 7) / 8, 256, 0, stream>>>(fb, Wb, x);

    // 4. scatter-add over edges: one wave per edge
    k_scatter<<<(N_EDGES * 32) / 256, 256, 0, stream>>>(x, src, dst, out);

    // 5. in-degree norm + bias + relu + dropout, in place on d_out
    k_finalize<<<4096, 256, 0, stream>>>(out, deg_in, bias);
}
